// KbitLoraModel_90503550861850
// MI455X (gfx1250) — matrix operand order repaired and checked by reading, not compiled, not run on hardware
//
#include <hip/hip_runtime.h>
#include <cstddef>

typedef __attribute__((ext_vector_type(16))) _Float16 v16h;
typedef __attribute__((ext_vector_type(8)))  float    v8f;

// ---- model constants (match reference) ----
#define Lc   2
#define Hc   1024
#define NHc  16
#define HDc  64
#define NKVc 4
#define KVDc 256
#define Ic   2816
#define Vc   32000
#define Rc   64
#define Sc   512
#define BLKc 64
#define LORA_Sc (16.0f / 64.0f)
#define EPSc 1e-6f
#define THETAc 10000.0f

// ============================================================
// WMMA fragment loaders (CDNA5 16x16x32 f16 layouts, wave32)
// ============================================================

// A matrix 16x32 (MxK), f32 source converted to f16.
// lanes 0-15 : row M=lane,    K = 0..15
// lanes16-31 : row M=lane-16, K = 16..31
__device__ __forceinline__ v16h load_a16(const float* __restrict__ X, int ldx,
                                         int row, int k0, int lane) {
  const int m  = row + (lane & 15);
  const int kh = (lane >> 4) * 16;
  const float* p = X + (size_t)m * ldx + k0;
  v16h a;
#pragma unroll
  for (int i = 0; i < 16; ++i) a[i] = (_Float16)p[kh + i];
  return a;
}

// B matrix 32x16 (KxN) from quantized row-major W[N,K]:
// lanes 0-15 : col N=lane,    K = 0..15
// lanes16-31 : col N=lane-16, K = 16..31
// BLK=64 absmax blocks along K: a 32-chunk at k0 (k0%32==0) never straddles one.
__device__ __forceinline__ v16h load_b16_quant(const int* __restrict__ idx,
                                               const float* __restrict__ am,
                                               const float* __restrict__ cb,
                                               int K, int col, int k0, int lane) {
  const int n  = col + (lane & 15);
  const int kb = (lane >> 4) * 16;
  const int* p  = idx + (size_t)n * K + k0 + kb;
  const float s = am[((size_t)n * K + (size_t)k0) >> 6];
  v16h b;
#pragma unroll
  for (int i = 0; i < 16; ++i) b[i] = (_Float16)(cb[p[i] & 15] * s);
  return b;
}

__device__ __forceinline__ v16h load_b16_f32(const float* __restrict__ W,
                                             int K, int col, int k0, int lane) {
  const int n  = col + (lane & 15);
  const int kb = (lane >> 4) * 16;
  const float* p = W + (size_t)n * K + k0 + kb;
  v16h b;
#pragma unroll
  for (int i = 0; i < 16; ++i) b[i] = (_Float16)p[i];
  return b;
}

__device__ __forceinline__ v8f wmma_f16(v16h a, v16h b, v8f c) {
  return __builtin_amdgcn_wmma_f32_16x16x32_f16(false, a, false, b,
                                                (short)0, c, false, false);
}

// ============================================================
// Big GEMM: Y[M,N] (+)= X[M,K] * deq(W[N,K])^T
// block tile 32x256, 4 waves; wave tile 32x64 = 2 (M) x 4 (N) WMMA frags
// grid = (N/256, M/32), block = 128. N%256==0, M%32==0, K%32==0 assumed.
// ============================================================

template <int ACC>
__global__ void __launch_bounds__(128)
k_gemm_quant_big(const float* __restrict__ X, const int* __restrict__ idx,
                 const float* __restrict__ am, const float* __restrict__ cbg,
                 float* __restrict__ Y, int N, int K) {
  __shared__ float cb[16];
  if (threadIdx.x < 16) cb[threadIdx.x] = cbg[threadIdx.x];
  __syncthreads();
  const int lane = threadIdx.x & 31;
  const int wave = threadIdx.x >> 5;
  const int row = blockIdx.y * 32;
  const int col = blockIdx.x * 256 + wave * 64;
  v8f c[2][4];
#pragma unroll
  for (int mi = 0; mi < 2; ++mi)
#pragma unroll
    for (int t = 0; t < 4; ++t)
#pragma unroll
      for (int j = 0; j < 8; ++j) c[mi][t][j] = 0.0f;

  for (int k0 = 0; k0 < K; k0 += 32) {
    if (k0 + 64 <= K)  // prefetch next index chunk (global_prefetch_b8)
      __builtin_prefetch(idx + (size_t)(col + (lane & 15)) * K + k0 + 64, 0, 1);
    const v16h a0 = load_a16(X, K, row, k0, lane);
    const v16h a1 = load_a16(X, K, row + 16, k0, lane);
#pragma unroll
    for (int t = 0; t < 4; ++t) {
      const v16h b = load_b16_quant(idx, am, cb, K, col + 16 * t, k0, lane);
      c[0][t] = wmma_f16(a0, b, c[0][t]);
      c[1][t] = wmma_f16(a1, b, c[1][t]);
    }
  }
  // C/D layout: elem j -> row = rowFrag + j + 8*(lane>>4), col = colFrag + (lane&15)
  const int nl = lane & 15;
  const int mh = (lane >> 4) * 8;
#pragma unroll
  for (int mi = 0; mi < 2; ++mi)
#pragma unroll
    for (int t = 0; t < 4; ++t) {
      const int m0 = row + mi * 16 + mh;
      const int n  = col + 16 * t + nl;
#pragma unroll
      for (int j = 0; j < 8; ++j) {
        float* dst = Y + (size_t)((unsigned)(m0 + j) * (unsigned)N + (unsigned)n);
        if (ACC) *dst += c[mi][t][j];
        else     *dst  = c[mi][t][j];
      }
    }
}

template <int ACC>
__global__ void __launch_bounds__(128)
k_gemm_f32_big(const float* __restrict__ X, const float* __restrict__ W,
               float* __restrict__ Y, int N, int K, float scale) {
  const int lane = threadIdx.x & 31;
  const int wave = threadIdx.x >> 5;
  const int row = blockIdx.y * 32;
  const int col = blockIdx.x * 256 + wave * 64;
  v8f c[2][4];
#pragma unroll
  for (int mi = 0; mi < 2; ++mi)
#pragma unroll
    for (int t = 0; t < 4; ++t)
#pragma unroll
      for (int j = 0; j < 8; ++j) c[mi][t][j] = 0.0f;

  for (int k0 = 0; k0 < K; k0 += 32) {
    const v16h a0 = load_a16(X, K, row, k0, lane);
    const v16h a1 = load_a16(X, K, row + 16, k0, lane);
#pragma unroll
    for (int t = 0; t < 4; ++t) {
      const v16h b = load_b16_f32(W, K, col + 16 * t, k0, lane);
      c[0][t] = wmma_f16(a0, b, c[0][t]);
      c[1][t] = wmma_f16(a1, b, c[1][t]);
    }
  }
  const int nl = lane & 15;
  const int mh = (lane >> 4) * 8;
#pragma unroll
  for (int mi = 0; mi < 2; ++mi)
#pragma unroll
    for (int t = 0; t < 4; ++t) {
      const int m0 = row + mi * 16 + mh;
      const int n  = col + 16 * t + nl;
#pragma unroll
      for (int j = 0; j < 8; ++j) {
        float* dst = Y + (size_t)((unsigned)(m0 + j) * (unsigned)N + (unsigned)n);
        if (ACC) *dst += scale * c[mi][t][j];
        else     *dst  = scale * c[mi][t][j];
      }
    }
}

// Small GEMM for LoRA stage 1 (N = 64): wave tile 16x16, block 64 cols.
// grid = (1, M/16), block = 128.
__global__ void __launch_bounds__(128)
k_gemm_f32_small(const float* __restrict__ X, const float* __restrict__ W,
                 float* __restrict__ Y, int N, int K) {
  const int lane = threadIdx.x & 31;
  const int wave = threadIdx.x >> 5;
  const int row = blockIdx.y * 16;
  const int col = blockIdx.x * 64 + wave * 16;
  v8f c;
#pragma unroll
  for (int j = 0; j < 8; ++j) c[j] = 0.0f;
  for (int k0 = 0; k0 < K; k0 += 32) {
    const v16h a = load_a16(X, K, row, k0, lane);
    const v16h b = load_b16_f32(W, K, col, k0, lane);
    c = wmma_f16(a, b, c);
  }
  const int n  = col + (lane & 15);
  const int m0 = row + (lane >> 4) * 8;
#pragma unroll
  for (int j = 0; j < 8; ++j)
    Y[(size_t)((unsigned)(m0 + j) * (unsigned)N + (unsigned)n)] = c[j];
}

// ============================================================
// Support kernels
// ============================================================

__global__ void k_embed(const int* __restrict__ ids, const float* __restrict__ emb,
                        float* __restrict__ h) {
  const int row = blockIdx.x;
  const int id  = ids[row];
  const float* src = emb + (size_t)id * Hc;
  float* dst = h + (size_t)row * Hc;
  for (int i = threadIdx.x; i < Hc; i += blockDim.x) dst[i] = src[i];
}

__global__ void __launch_bounds__(256)
k_rmsnorm(const float* __restrict__ h, const float* __restrict__ w,
          float* __restrict__ out) {
  __shared__ float red[256];
  const int row = blockIdx.x;
  const float* x = h + (size_t)row * Hc;
  float s = 0.f;
  for (int i = threadIdx.x; i < Hc; i += 256) { float v = x[i]; s += v * v; }
  red[threadIdx.x] = s;
  __syncthreads();
  for (int st = 128; st > 0; st >>= 1) {
    if (threadIdx.x < st) red[threadIdx.x] += red[threadIdx.x + st];
    __syncthreads();
  }
  const float rs = rsqrtf(red[0] / (float)Hc + EPSc);
  for (int i = threadIdx.x; i < Hc; i += 256)
    out[(size_t)row * Hc + i] = x[i] * rs * w[i];
}

// in-place llama rotate_half RoPE; grid=(S, nheads), block=32
__global__ void __launch_bounds__(32)
k_rope(float* __restrict__ t, int nheads) {
  const int s  = blockIdx.x;
  const int hh = blockIdx.y;
  float* p = t + ((size_t)s * nheads + hh) * HDc;
  const int d = threadIdx.x;  // 0..31
  const float f   = __expf(-__logf(THETAc) * (2.0f * (float)d) / (float)HDc);
  const float ang = (float)s * f;
  const float cc = __cosf(ang), sn = __sinf(ang);
  const float x1 = p[d], x2 = p[d + 32];
  p[d]      = x1 * cc - x2 * sn;
  p[d + 32] = x2 * cc + x1 * sn;
}

// causal softmax attention, GQA; grid=(S, NH), block=128
__global__ void __launch_bounds__(128)
k_attn(const float* __restrict__ q, const float* __restrict__ k,
       const float* __restrict__ v, float* __restrict__ ctx) {
  __shared__ float qs[HDc];
  __shared__ float sc[Sc];
  __shared__ float red[128];
  const int qpos = blockIdx.x;
  const int h    = blockIdx.y;
  const int kvh  = h / (NHc / NKVc);
  const int tid  = threadIdx.x;
  const float* qv = q + ((size_t)qpos * NHc + h) * HDc;
  if (tid < HDc) qs[tid] = qv[tid];
  __syncthreads();
  float lmax = -1e30f;
  for (int kk = tid; kk <= qpos; kk += 128) {
    const float* kr = k + ((size_t)kk * NKVc + kvh) * HDc;
    float s = 0.f;
#pragma unroll 8
    for (int d = 0; d < HDc; ++d) s += qs[d] * kr[d];
    s *= 0.125f;  // 1/sqrt(64)
    sc[kk] = s;
    lmax = fmaxf(lmax, s);
  }
  red[tid] = lmax; __syncthreads();
  for (int st = 64; st > 0; st >>= 1) {
    if (tid < st) red[tid] = fmaxf(red[tid], red[tid + st]);
    __syncthreads();
  }
  const float mx = red[0];
  __syncthreads();
  float lsum = 0.f;
  for (int kk = tid; kk <= qpos; kk += 128) {
    float e = __expf(sc[kk] - mx);
    sc[kk] = e;
    lsum += e;
  }
  red[tid] = lsum; __syncthreads();
  for (int st = 64; st > 0; st >>= 1) {
    if (tid < st) red[tid] += red[tid + st];
    __syncthreads();
  }
  const float inv = 1.0f / red[0];
  __syncthreads();
  for (int d = tid; d < HDc; d += 128) {
    float acc = 0.f;
    for (int kk = 0; kk <= qpos; ++kk)
      acc += sc[kk] * v[((size_t)kk * NKVc + kvh) * HDc + d];
    ctx[((size_t)qpos * NHc + h) * HDc + d] = acc * inv;
  }
}

__global__ void k_silu_mul(const float* __restrict__ g, const float* __restrict__ u,
                           float* __restrict__ o, int n) {
  const int i = blockIdx.x * blockDim.x + threadIdx.x;
  if (i < n) {
    float gv = g[i];
    o[i] = (gv / (1.f + __expf(-gv))) * u[i];
  }
}

// ============================================================
// host-side launch helpers
// ============================================================

static inline void gemm_q(const float* X, const int* idx, const float* am,
                          const float* cb, float* Y, int N, int K,
                          int acc, hipStream_t s) {
  dim3 g(N / 256, Sc / 32);
  if (acc) k_gemm_quant_big<1><<<g, 128, 0, s>>>(X, idx, am, cb, Y, N, K);
  else     k_gemm_quant_big<0><<<g, 128, 0, s>>>(X, idx, am, cb, Y, N, K);
}

static inline void gemm_f_big(const float* X, const float* W, float* Y,
                              int N, int K, float scale, int acc, hipStream_t s) {
  dim3 g(N / 256, Sc / 32);
  if (acc) k_gemm_f32_big<1><<<g, 128, 0, s>>>(X, W, Y, N, K, scale);
  else     k_gemm_f32_big<0><<<g, 128, 0, s>>>(X, W, Y, N, K, scale);
}

// quant linear with LoRA: Y (+)= X*deq(W)^T + LORA_S*(X*A^T)*B^T
static inline void linear_lora(const float* X, const int* idx, const float* am,
                               const float* cb, const float* A, const float* Bm,
                               float* t, float* Y, int N, int K, int acc,
                               hipStream_t s) {
  gemm_q(X, idx, am, cb, Y, N, K, acc, s);
  k_gemm_f32_small<<<dim3(1, Sc / 16), 128, 0, s>>>(X, A, t, Rc, K);
  gemm_f_big(t, Bm, Y, N, Rc, LORA_Sc, 1, s);
}

extern "C" void kernel_launch(void* const* d_in, const int* in_sizes, int n_in,
                              void* d_out, int out_size, void* d_ws, size_t ws_size,
                              hipStream_t stream) {
  (void)in_sizes; (void)n_in; (void)out_size; (void)ws_size;
  // inputs in setup_inputs() order
  const int*   input_ids = (const int*)d_in[0];
  const float* embed     = (const float*)d_in[1];
  const float* codebook  = (const float*)d_in[2];
  const int*   q_idx = (const int*)d_in[3];  const float* q_am = (const float*)d_in[4];
  const float* qA = (const float*)d_in[5];   const float* qB = (const float*)d_in[6];
  const int*   k_idx = (const int*)d_in[7];  const float* k_am = (const float*)d_in[8];
  const float* kA = (const float*)d_in[9];   const float* kB = (const float*)d_in[10];
  const int*   v_idx = (const int*)d_in[11]; const float* v_am = (const float*)d_in[12];
  const float* vA = (const float*)d_in[13];  const float* vB = (const float*)d_in[14];
  const int*   o_idx = (const int*)d_in[15]; const float* o_am = (const float*)d_in[16];
  const float* oA = (const float*)d_in[17];  const float* oB = (const float*)d_in[18];
  const int*   g_idx = (const int*)d_in[19]; const float* g_am = (const float*)d_in[20];
  const float* gA = (const float*)d_in[21];  const float* gB = (const float*)d_in[22];
  const int*   u_idx = (const int*)d_in[23]; const float* u_am = (const float*)d_in[24];
  const float* uA = (const float*)d_in[25];  const float* uB = (const float*)d_in[26];
  const int*   dd_idx = (const int*)d_in[27]; const float* dd_am = (const float*)d_in[28];
  const float* dA = (const float*)d_in[29];  const float* dB = (const float*)d_in[30];
  const float* ln1 = (const float*)d_in[31];
  const float* ln2 = (const float*)d_in[32];
  const float* final_norm = (const float*)d_in[33];
  const int*   lm_idx = (const int*)d_in[34];
  const float* lm_am  = (const float*)d_in[35];

  // workspace carving (all f32)
  float* h   = (float*)d_ws;          // [S,H]
  float* x   = h  + (size_t)Sc * Hc;  // [S,H]
  float* qb  = x  + (size_t)Sc * Hc;  // [S,H]
  float* kb  = qb + (size_t)Sc * Hc;  // [S,KVD]
  float* vb  = kb + (size_t)Sc * KVDc;// [S,KVD]
  float* cx  = vb + (size_t)Sc * KVDc;// [S,H]
  float* tb  = cx + (size_t)Sc * Hc;  // [S,R]
  float* gb  = tb + (size_t)Sc * Rc;  // [S,I]
  float* ub  = gb + (size_t)Sc * Ic;  // [S,I]
  float* ab  = ub + (size_t)Sc * Ic;  // [S,I]

  k_embed<<<Sc, 256, 0, stream>>>(input_ids, embed, h);

  for (int l = 0; l < Lc; ++l) {
    // ---- attention block ----
    k_rmsnorm<<<Sc, 256, 0, stream>>>(h, ln1 + (size_t)l * Hc, x);
    linear_lora(x, q_idx + (size_t)l * Hc * Hc,  q_am + (size_t)l * (Hc * Hc / BLKc),
                codebook, qA + (size_t)l * Rc * Hc, qB + (size_t)l * Hc * Rc,
                tb, qb, Hc, Hc, 0, stream);
    linear_lora(x, k_idx + (size_t)l * KVDc * Hc, k_am + (size_t)l * (KVDc * Hc / BLKc),
                codebook, kA + (size_t)l * Rc * Hc, kB + (size_t)l * KVDc * Rc,
                tb, kb, KVDc, Hc, 0, stream);
    linear_lora(x, v_idx + (size_t)l * KVDc * Hc, v_am + (size_t)l * (KVDc * Hc / BLKc),
                codebook, vA + (size_t)l * Rc * Hc, vB + (size_t)l * KVDc * Rc,
                tb, vb, KVDc, Hc, 0, stream);
    k_rope<<<dim3(Sc, NHc),  32, 0, stream>>>(qb, NHc);
    k_rope<<<dim3(Sc, NKVc), 32, 0, stream>>>(kb, NKVc);
    k_attn<<<dim3(Sc, NHc), 128, 0, stream>>>(qb, kb, vb, cx);
    // o-projection accumulates into residual h
    linear_lora(cx, o_idx + (size_t)l * Hc * Hc, o_am + (size_t)l * (Hc * Hc / BLKc),
                codebook, oA + (size_t)l * Rc * Hc, oB + (size_t)l * Hc * Rc,
                tb, h, Hc, Hc, 1, stream);
    // ---- SwiGLU MLP block ----
    k_rmsnorm<<<Sc, 256, 0, stream>>>(h, ln2 + (size_t)l * Hc, x);
    linear_lora(x, g_idx + (size_t)l * Ic * Hc, g_am + (size_t)l * (Ic * Hc / BLKc),
                codebook, gA + (size_t)l * Rc * Hc, gB + (size_t)l * Ic * Rc,
                tb, gb, Ic, Hc, 0, stream);
    linear_lora(x, u_idx + (size_t)l * Ic * Hc, u_am + (size_t)l * (Ic * Hc / BLKc),
                codebook, uA + (size_t)l * Rc * Hc, uB + (size_t)l * Ic * Rc,
                tb, ub, Ic, Hc, 0, stream);
    k_silu_mul<<<(Sc * Ic + 255) / 256, 256, 0, stream>>>(gb, ub, ab, Sc * Ic);
    // down-projection accumulates into residual h
    linear_lora(ab, dd_idx + (size_t)l * Hc * Ic, dd_am + (size_t)l * (Hc * Ic / BLKc),
                codebook, dA + (size_t)l * Rc * Ic, dB + (size_t)l * Hc * Rc,
                tb, h, Hc, Ic, 1, stream);
  }

  k_rmsnorm<<<Sc, 256, 0, stream>>>(h, final_norm, x);
  // quantized lm_head (no LoRA) straight into d_out: [S, V] f32 logits
  gemm_q(x, lm_idx, lm_am, codebook, (float*)d_out, Vc, Hc, 0, stream);
}
